// GGNNObj_5669356831746
// MI455X (gfx1250) — matrix operand order repaired and checked by reading, not compiled, not run on hardware
//
#include <hip/hip_runtime.h>
#include <hip/hip_bf16.h>
#include <math.h>

// ---------------------------------------------------------------------------
// GGNN object classifier on MI455X (gfx1250): all big linears via
// v_wmma_f32_16x16x32_f16 (f16 operands, f32 accumulate). B tiles staged
// into double-buffered LDS via GLOBAL_LOAD_ASYNC_TO_LDS_B128 (ASYNCcnt),
// with a sync fallback. Class-mixing einsum restructured to tiny FP32 VALU
// kernels.
// ---------------------------------------------------------------------------

#define NOBJ 256
#define CNUM 151
#define DH   512
#define MROW (NOBJ * CNUM)   // 38656 = 128 * 302
#define TSTEP 3

typedef __attribute__((ext_vector_type(16))) _Float16 v16h;
typedef __attribute__((ext_vector_type(8)))  float    v8f;
typedef int ggnn_v4i __attribute__((vector_size(16)));   // matches builtin param

struct alignas(16) h8 { _Float16 v[8]; };

#if defined(__has_builtin)
#if __has_builtin(__builtin_amdgcn_global_load_async_to_lds_b128)
#define GGNN_USE_ASYNC_LDS 1
#endif
#endif

__device__ inline void ggnn_wait_async() {
#ifdef GGNN_USE_ASYNC_LDS
#if __has_builtin(__builtin_amdgcn_s_wait_asynccnt)
  __builtin_amdgcn_s_wait_asynccnt(0);
#else
  asm volatile("s_wait_asynccnt 0x0" ::: "memory");
#endif
#endif
}

// ---- fragment loaders ------------------------------------------------------
// A (16-bit, 16x32, dense table): lanes 0-15 -> rows, halves 0..7 = K(base..+7),
// halves 8..15 = K(16+base..), base = 0 (lanes 0-15) or 8 (lanes 16-31).
__device__ inline v16h load_frag_a(const _Float16* __restrict__ A,
                                   long long rowTimesK, int k0, int lane) {
  const int base = (lane < 16) ? 0 : 8;
  const _Float16* p = A + rowTimesK + k0 + base;
  const h8* lo = (const h8*)p;        // 16B aligned (K mult of 8, k0 mult of 32)
  const h8* hi = (const h8*)(p + 16);
  v16h r;
#pragma unroll
  for (int i = 0; i < 8; ++i) { r[i] = lo->v[i]; r[8 + i] = hi->v[i]; }
  return r;
}

// B (16-bit, 32x16, column-major per lane): lane -> col = lane&15,
// halves 0..7 = K(base..+7), halves 8..15 = K(16+base..), base 0 (lanes 0-15) / 8 (lanes 16-31).
// LDS tile layout: lds[col*32 + k], 64 cols x 32 k halves.
__device__ inline v16h load_frag_b(const _Float16* __restrict__ lds,
                                   int c0, int lane) {
  const int col = c0 + (lane & 15);
  const int kb  = (lane < 16) ? 0 : 8;
  const _Float16* p = lds + col * 32 + kb;
  const h8* q0 = (const h8*)p;
  const h8* q1 = (const h8*)(p + 16);
  v16h r;
#pragma unroll
  for (int i = 0; i < 8; ++i) { r[i] = q0->v[i]; r[8 + i] = q1->v[i]; }
  return r;
}

// ---- WMMA GEMM: OUT[M,nvalid] = A[M,K] @ W[nvalid,K]^T + bias (+add) (+ep) -
// block = 256 threads (8 waves); block tile 128 rows x 64 cols; wave 16x64.
// ep: 0 none, 1 sigmoid, 2 tanh, 3 relu.  grid = (ceil(nvalid/64), Mrows/128).
__global__ __launch_bounds__(256) void gemm16(
    const _Float16* __restrict__ A, const _Float16* __restrict__ W,
    const float* __restrict__ bias, const float* __restrict__ add,
    float* __restrict__ outF, _Float16* __restrict__ outH,
    int K, int ldo, int nvalid, int ep) {
  __shared__ _Float16 ldsB[2][64 * 32];   // double-buffered 4 KB B tiles

  const int lane = threadIdx.x & 31;
  const int wid  = threadIdx.x >> 5;
  const int colBlock = blockIdx.x * 64;
  const long long rowWave = (long long)blockIdx.y * 128 + wid * 16;

  v8f acc[4];
#pragma unroll
  for (int f = 0; f < 4; ++f)
#pragma unroll
    for (int j = 0; j < 8; ++j) acc[f][j] = 0.0f;

  // B staging: 256 threads x 16B = 4 KB chunk. thread -> (col, 8-half segment)
  const int stageCol = threadIdx.x >> 2;          // 0..63
  const int stageOff = (threadIdx.x & 3) * 8;     // 0,8,16,24 halves
  const int gcol = colBlock + stageCol;
  const bool colOk = (gcol < nvalid);
  const _Float16* wrow = W + (size_t)gcol * (size_t)K;

  const long long rowTimesK = (rowWave + (lane & 15)) * (long long)K;

  auto stage = [&](int k0, int buf) {
    _Float16* dst = &ldsB[buf][stageCol * 32 + stageOff];
    if (colOk) {
#ifdef GGNN_USE_ASYNC_LDS
      __builtin_amdgcn_global_load_async_to_lds_b128(
          (__attribute__((address_space(1))) ggnn_v4i*)(wrow + k0 + stageOff),
          (__attribute__((address_space(3))) ggnn_v4i*)dst, 0, 0);
#else
      *(h8*)dst = *(const h8*)(wrow + k0 + stageOff);
#endif
    } else {
      h8 z;
#pragma unroll
      for (int i = 0; i < 8; ++i) z.v[i] = (_Float16)0.0f;
      *(h8*)dst = z;
    }
  };

  // preload first chunk
  stage(0, 0);
  ggnn_wait_async();
  __syncthreads();

  int cur = 0;
  for (int k0 = 0; k0 < K; k0 += 32) {
    const int nk = k0 + 32;
    if (nk < K) stage(nk, cur ^ 1);                 // overlap with compute
    if (nk + 32 < K)                                 // prefetch A two chunks out
      __builtin_prefetch((const void*)(A + rowTimesK + nk + 32), 0, 0);

    v16h a = load_frag_a(A, rowTimesK, k0, lane);
    const _Float16* bb = ldsB[cur];
    v16h b0 = load_frag_b(bb, 0,  lane);
    v16h b1 = load_frag_b(bb, 16, lane);
    v16h b2 = load_frag_b(bb, 32, lane);
    v16h b3 = load_frag_b(bb, 48, lane);
    acc[0] = __builtin_amdgcn_wmma_f32_16x16x32_f16(false, a, false, b0,
                                                    (short)0, acc[0], false, false);
    acc[1] = __builtin_amdgcn_wmma_f32_16x16x32_f16(false, a, false, b1,
                                                    (short)0, acc[1], false, false);
    acc[2] = __builtin_amdgcn_wmma_f32_16x16x32_f16(false, a, false, b2,
                                                    (short)0, acc[2], false, false);
    acc[3] = __builtin_amdgcn_wmma_f32_16x16x32_f16(false, a, false, b3,
                                                    (short)0, acc[3], false, false);

    ggnn_wait_async();    // own async LDS writes landed
    __syncthreads();      // whole block's stage of next buffer visible
    cur ^= 1;
  }

  // epilogue: C/D layout -> lane col = lane&15, vgpr j row = j + (lane<16?0:8)
#pragma unroll
  for (int f = 0; f < 4; ++f) {
    const int col = colBlock + f * 16 + (lane & 15);
    if (col < nvalid) {
      const float bv = bias ? bias[col] : 0.0f;
#pragma unroll
      for (int j = 0; j < 8; ++j) {
        const long long row = rowWave + j + ((lane < 16) ? 0 : 8);
        float v = acc[f][j] + bv;
        if (add) v += add[row * ldo + col];
        if (ep == 1)      v = 1.0f / (1.0f + expf(-v));
        else if (ep == 2) v = tanhf(v);
        else if (ep == 3) v = fmaxf(v, 0.0f);
        if (outF) outF[row * ldo + col] = v;
        if (outH) outH[row * ldo + col] = (_Float16)v;
      }
    }
  }
}

// ---- small helper kernels --------------------------------------------------
__global__ void k_f32_to_f16(const float* __restrict__ s, _Float16* __restrict__ d,
                             long long n) {
  long long i = (long long)blockIdx.x * blockDim.x + threadIdx.x;
  long long stride = (long long)gridDim.x * blockDim.x;
  for (; i < n; i += stride) d[i] = (_Float16)s[i];
}

__global__ void k_init_hidden(const float* __restrict__ inp,
                              float* __restrict__ h, _Float16* __restrict__ hh) {
  long long idx = (long long)blockIdx.x * blockDim.x + threadIdx.x;
  if (idx >= (long long)MROW * DH) return;
  int r = (int)(idx >> 9), dch = (int)(idx & 511);
  int n = r / CNUM;
  float v = inp[n * DH + dch];
  h[idx] = v; hh[idx] = (_Float16)v;
}

__global__ void k_hidden_sum(const float* __restrict__ h, float* __restrict__ hs) {
  int idx = blockIdx.x * blockDim.x + threadIdx.x;
  if (idx >= CNUM * DH) return;
  float s = 0.0f;
  for (int n = 0; n < NOBJ; ++n) s += h[(long long)n * (CNUM * DH) + idx];
  hs[idx] = s;
}

__global__ void k_msum(const float* __restrict__ mat, const float* __restrict__ hs,
                       float* __restrict__ ms1, float* __restrict__ ms2) {
  int idx = blockIdx.x * blockDim.x + threadIdx.x;
  if (idx >= CNUM * DH) return;
  int c = idx >> 9, dch = idx & 511;
  float s1 = 0.0f, s2 = 0.0f;
  for (int x = 0; x < CNUM; ++x) {
    float hv = hs[x * DH + dch];
    s1 += mat[x * CNUM + c] * hv;   // matrix^T @ hs
    s2 += mat[c * CNUM + x] * hv;   // matrix   @ hs
  }
  ms1[idx] = s1; ms2[idx] = s2;
}

// av1[n,c,d] = ms1[c,d] - (matrix^T @ h[n])[c,d];  av2 analogous.
__global__ void k_av(const float* __restrict__ mat, const float* __restrict__ h,
                     const float* __restrict__ ms1, const float* __restrict__ ms2,
                     _Float16* __restrict__ av) {
  long long idx = (long long)blockIdx.x * blockDim.x + threadIdx.x;
  if (idx >= (long long)MROW * DH) return;
  int r = (int)(idx >> 9), dch = (int)(idx & 511);
  int n = r / CNUM, c = r % CNUM;
  float s1 = ms1[c * DH + dch], s2 = ms2[c * DH + dch];
  const float* hb = h + (long long)n * (CNUM * DH) + dch;
  for (int x = 0; x < CNUM; ++x) {
    float hv = hb[(long long)x * DH];
    s1 -= mat[x * CNUM + c] * hv;
    s2 -= mat[c * CNUM + x] * hv;
  }
  av[(long long)r * (2 * DH) + dch]      = (_Float16)s1;
  av[(long long)r * (2 * DH) + DH + dch] = (_Float16)s2;
}

__global__ void k_mulrh(const float* __restrict__ rv, const float* __restrict__ h,
                        _Float16* __restrict__ rh) {
  long long idx = (long long)blockIdx.x * blockDim.x + threadIdx.x;
  if (idx >= (long long)MROW * DH) return;
  rh[idx] = (_Float16)(rv[idx] * h[idx]);
}

__global__ void k_update(const float* __restrict__ zv, const float* __restrict__ hv,
                         float* __restrict__ h, _Float16* __restrict__ hh) {
  long long idx = (long long)blockIdx.x * blockDim.x + threadIdx.x;
  if (idx >= (long long)MROW * DH) return;
  float z = zv[idx];
  float nh = (1.0f - z) * h[idx] + z * hv[idx];
  h[idx] = nh; hh[idx] = (_Float16)nh;
}

__global__ void k_concat(const _Float16* __restrict__ hh,
                         const _Float16* __restrict__ inh,
                         _Float16* __restrict__ cat) {
  long long idx = (long long)blockIdx.x * blockDim.x + threadIdx.x;
  if (idx >= (long long)MROW * DH) return;
  int r = (int)(idx >> 9), dch = (int)(idx & 511);
  int n = r / CNUM;
  cat[(long long)r * (2 * DH) + dch]      = hh[idx];
  cat[(long long)r * (2 * DH) + DH + dch] = inh[n * DH + dch];
}

// ---------------------------------------------------------------------------
extern "C" void kernel_launch(void* const* d_in, const int* in_sizes, int n_in,
                              void* d_out, int out_size, void* d_ws, size_t ws_size,
                              hipStream_t stream) {
  (void)in_sizes; (void)n_in; (void)out_size; (void)ws_size;
  const float* input = (const float*)d_in[0];
  const float* mat   = (const float*)d_in[1];
  const float* w3w = (const float*)d_in[2];  const float* b3w = (const float*)d_in[3];
  const float* w3u = (const float*)d_in[4];  const float* b3u = (const float*)d_in[5];
  const float* w4w = (const float*)d_in[6];  const float* b4w = (const float*)d_in[7];
  /* w4u (8), b4u (9) intentionally unused (faithful to reference bug) */
  const float* w5w = (const float*)d_in[10]; const float* b5w = (const float*)d_in[11];
  const float* w5u = (const float*)d_in[12]; const float* b5u = (const float*)d_in[13];
  const float* wo  = (const float*)d_in[14]; const float* bo  = (const float*)d_in[15];
  const float* wc  = (const float*)d_in[16]; const float* bc  = (const float*)d_in[17];
  float* out = (float*)d_out;

  // ---- workspace carve ----
  char* p = (char*)d_ws;
  auto carve = [&](size_t bytes) -> char* {
    char* r = p; p += (bytes + 255) & ~(size_t)255; return r;
  };
  const size_t MD4 = (size_t)MROW * DH * 4, MD2 = (size_t)MROW * DH * 2;
  float*    h_f32 = (float*)   carve(MD4);
  float*    bufU  = (float*)   carve(MD4);
  float*    bufZ  = (float*)   carve(MD4);
  float*    bufR  = (float*)   carve(MD4);
  _Float16* av16  = (_Float16*)carve((size_t)MROW * 2 * DH * 2);
  _Float16* h16   = (_Float16*)carve(MD2);
  _Float16* rh16  = (_Float16*)carve(MD2);
  _Float16* in16  = (_Float16*)carve((size_t)NOBJ * DH * 2);
  float*    hs    = (float*)   carve((size_t)CNUM * DH * 4);
  float*    ms1   = (float*)   carve((size_t)CNUM * DH * 4);
  float*    ms2   = (float*)   carve((size_t)CNUM * DH * 4);
  _Float16* w3w16 = (_Float16*)carve((size_t)DH * 2 * DH * 2);
  _Float16* w4w16 = (_Float16*)carve((size_t)DH * 2 * DH * 2);
  _Float16* w5w16 = (_Float16*)carve((size_t)DH * 2 * DH * 2);
  _Float16* wo16  = (_Float16*)carve((size_t)DH * 2 * DH * 2);
  _Float16* w3u16 = (_Float16*)carve((size_t)DH * DH * 2);
  _Float16* w5u16 = (_Float16*)carve((size_t)DH * DH * 2);
  _Float16* wc16  = (_Float16*)carve((size_t)CNUM * CNUM * DH * 2);

  const int TPB = 256;
  auto blocksFor = [](long long n) { return (unsigned)((n + 255) / 256); };

  // ---- weights / input -> f16 ----
  k_f32_to_f16<<<blocksFor(DH * 2 * DH), TPB, 0, stream>>>(w3w, w3w16, DH * 2 * DH);
  k_f32_to_f16<<<blocksFor(DH * 2 * DH), TPB, 0, stream>>>(w4w, w4w16, DH * 2 * DH);
  k_f32_to_f16<<<blocksFor(DH * 2 * DH), TPB, 0, stream>>>(w5w, w5w16, DH * 2 * DH);
  k_f32_to_f16<<<blocksFor(DH * 2 * DH), TPB, 0, stream>>>(wo,  wo16,  DH * 2 * DH);
  k_f32_to_f16<<<blocksFor(DH * DH),     TPB, 0, stream>>>(w3u, w3u16, DH * DH);
  k_f32_to_f16<<<blocksFor(DH * DH),     TPB, 0, stream>>>(w5u, w5u16, DH * DH);
  k_f32_to_f16<<<blocksFor((long long)CNUM * CNUM * DH), TPB, 0, stream>>>(
      wc, wc16, (long long)CNUM * CNUM * DH);
  k_f32_to_f16<<<blocksFor(NOBJ * DH),   TPB, 0, stream>>>(input, in16, NOBJ * DH);

  // ---- init hidden ----
  const long long MD = (long long)MROW * DH;
  k_init_hidden<<<blocksFor(MD), TPB, 0, stream>>>(input, h_f32, h16);

  dim3 blk(256);
  dim3 gMain(DH / 64, MROW / 128);    // (8, 302)

  for (int t = 0; t < TSTEP; ++t) {
    k_hidden_sum<<<blocksFor(CNUM * DH), TPB, 0, stream>>>(h_f32, hs);
    k_msum<<<blocksFor(CNUM * DH), TPB, 0, stream>>>(mat, hs, ms1, ms2);
    k_av<<<blocksFor(MD), TPB, 0, stream>>>(mat, h_f32, ms1, ms2, av16);

    // shared term: bufU = h @ w3u^T + b3u
    gemm16<<<gMain, blk, 0, stream>>>(h16, w3u16, b3u, nullptr, bufU, nullptr,
                                      DH, DH, DH, 0);
    // zv = sigmoid(av @ w3w^T + b3w + bufU)
    gemm16<<<gMain, blk, 0, stream>>>(av16, w3w16, b3w, bufU, bufZ, nullptr,
                                      2 * DH, DH, DH, 1);
    // rv = sigmoid(av @ w4w^T + b4w + bufU)   (reference reuses w3u/b3u term)
    gemm16<<<gMain, blk, 0, stream>>>(av16, w4w16, b4w, bufU, bufR, nullptr,
                                      2 * DH, DH, DH, 1);
    // rh = rv * h
    k_mulrh<<<blocksFor(MD), TPB, 0, stream>>>(bufR, h_f32, rh16);
    // bufU = rh @ w5u^T + b5u
    gemm16<<<gMain, blk, 0, stream>>>(rh16, w5u16, b5u, nullptr, bufU, nullptr,
                                      DH, DH, DH, 0);
    // hv = tanh(av @ w5w^T + b5w + bufU)  -> bufR
    gemm16<<<gMain, blk, 0, stream>>>(av16, w5w16, b5w, bufU, bufR, nullptr,
                                      2 * DH, DH, DH, 2);
    // h = (1-zv)*h + zv*hv
    k_update<<<blocksFor(MD), TPB, 0, stream>>>(bufZ, bufR, h_f32, h16);
  }

  // cat = [hidden | input_rep]  (into av16)
  k_concat<<<blocksFor(MD), TPB, 0, stream>>>(h16, in16, av16);
  // o = relu(cat @ wo^T + bo) -> f16, stored into h16 ([256, C*D] contiguous)
  gemm16<<<gMain, blk, 0, stream>>>(av16, wo16, bo, nullptr, nullptr, h16,
                                    2 * DH, DH, DH, 3);
  // logits = o @ wc^T + bc  ([256,151])
  dim3 gCls((CNUM + 63) / 64, NOBJ / 128);  // (3, 2)
  gemm16<<<gCls, blk, 0, stream>>>(h16, wc16, bc, nullptr, out, nullptr,
                                   CNUM * DH, CNUM, CNUM, 0);
}
